// HGT_56152402427957
// MI455X (gfx1250) — compile-verified, matrix operand order checked by reading, not run on hardware
//
#include <hip/hip_runtime.h>
#include <hip/hip_bf16.h>
#include <math.h>

// ---------------- types ----------------
typedef __attribute__((ext_vector_type(16))) __bf16 v16bf;
typedef __attribute__((ext_vector_type(8)))  __bf16 v8bf;
typedef __attribute__((ext_vector_type(8)))  float  v8f;

// ---------------- problem constants ----------------
#define NP   200000
#define NA   100000
#define NKW  50000
#define NTOT 350000
#define C_DIM 128
#define H_DIM 4
#define D_DIM 32
#define L_DIM 2
#define R_DIM 4
#define ETOT 600000
#define INV_SQRT_D 0.17677669529663687f

// ---------------- GEMM tiling ----------------
#define BM 128
#define BN 128
#define BK 32
#define LDS_STRIDE 40   // bf16 elements; 80 bytes/row keeps octets 16B-aligned

// epilogue modes
#define EPI_BIAS 0
#define EPI_RELU 1
#define EPI_SKIP 2

__device__ __forceinline__ float gelu_exact(float v)
{
  return 0.5f * v * (1.0f + erff(v * 0.70710678118654752f));
}

// -------------------------------------------------------------------
// Tiled GEMM: Cout[M,Nw] = epi( actA(A[M,Kd]) @ W[Kd,Nw] + bias )
// A fp32 (optionally exact-GELU'd on load), W bf16 (preconverted),
// fp32 accumulation via v_wmma_f32_16x16x32_bf16.
// 256 threads = 8 waves; each wave owns 16 rows x 128 cols (8 wmma accums).
// All 8 B fragments are preloaded so the 8 WMMAs issue under one DS wait.
// -------------------------------------------------------------------
__global__ __launch_bounds__(256) void gemm_bf16_wmma(
    const float* __restrict__ A, const __bf16* __restrict__ W,
    const float* __restrict__ bias, float* __restrict__ Cout,
    const float* __restrict__ Xprev, const float* __restrict__ skipgate,
    int M, int Nw, int Kd, int actA, int epi)
{
  __shared__ __align__(16) __bf16 As[BM * LDS_STRIDE];
  __shared__ __align__(16) __bf16 Bs[BN * LDS_STRIDE];

  const int tid  = threadIdx.x;
  const int m0   = blockIdx.x * BM;
  const int n0   = blockIdx.y * BN;
  const int wave = tid >> 5;
  const int lane = tid & 31;
  const int lh   = lane & 15;   // row/col within 16-wide tile
  const int kh   = lane >> 4;   // selects K-octet half per ISA bf16 layout

  v8f acc[8];
  #pragma unroll
  for (int i = 0; i < 8; ++i) acc[i] = (v8f){};

  for (int k0 = 0; k0 < Kd; k0 += BK) {
    // ---- stage A tile: float4 x2 load, optional GELU, packed b128 LDS store ----
    {
      const int r  = tid >> 2;        // 0..63
      const int kp = (tid & 3) * 8;   // K octet
      #pragma unroll
      for (int it = 0; it < 2; ++it) {
        const int rr = r + it * 64;
        const int gr = m0 + rr;
        const float* ap = A + (size_t)gr * Kd + k0 + kp;
        float4 f0 = make_float4(0.f, 0.f, 0.f, 0.f);
        float4 f1 = make_float4(0.f, 0.f, 0.f, 0.f);
        if (gr < M) {
          f0 = *(const float4*)ap;
          f1 = *(const float4*)(ap + 4);
          if (k0 + BK < Kd) __builtin_prefetch(ap + BK, 0, 3);  // WGP-scope prefetch
        }
        float vv[8] = {f0.x, f0.y, f0.z, f0.w, f1.x, f1.y, f1.z, f1.w};
        v8bf pk;
        #pragma unroll
        for (int i = 0; i < 8; ++i) {
          float v = vv[i];
          if (actA) v = gelu_exact(v);
          pk[i] = (__bf16)v;
        }
        *(v8bf*)&As[rr * LDS_STRIDE + kp] = pk;
      }
    }
    // ---- stage B tile transposed to Bs[n][k]; packed b128 LDS stores ----
    {
      const int n     = tid & 127;
      const int khalf = tid >> 7;     // 0 or 1 -> K 0..15 / 16..31
      const int nn    = n0 + n;
      const __bf16* wp = W + (size_t)(k0 + khalf * 16) * Nw + nn;
      v8bf p0, p1;
      #pragma unroll
      for (int i = 0; i < 8; ++i) {
        p0[i] = (nn < Nw) ? wp[(size_t)i * Nw]       : (__bf16)0.0f;
        p1[i] = (nn < Nw) ? wp[(size_t)(i + 8) * Nw] : (__bf16)0.0f;
      }
      *(v8bf*)&Bs[n * LDS_STRIDE + khalf * 16]     = p0;
      *(v8bf*)&Bs[n * LDS_STRIDE + khalf * 16 + 8] = p1;
    }
    __syncthreads();

    // ---- A fragment: lane holds row (wave*16+lh); K octets per ISA layout ----
    const int mr = wave * 16 + lh;
    v8bf alo = *(const v8bf*)&As[mr * LDS_STRIDE + kh * 8];
    v8bf ahi = *(const v8bf*)&As[mr * LDS_STRIDE + 16 + kh * 8];
    v16bf af = __builtin_shufflevector(alo, ahi, 0,1,2,3,4,5,6,7,8,9,10,11,12,13,14,15);

    // ---- preload all 8 B fragments, then issue the WMMA chain ----
    v16bf bfrag[8];
    #pragma unroll
    for (int nt = 0; nt < 8; ++nt) {
      const int nc = nt * 16 + lh;
      v8bf blo = *(const v8bf*)&Bs[nc * LDS_STRIDE + kh * 8];
      v8bf bhi = *(const v8bf*)&Bs[nc * LDS_STRIDE + 16 + kh * 8];
      bfrag[nt] = __builtin_shufflevector(blo, bhi, 0,1,2,3,4,5,6,7,8,9,10,11,12,13,14,15);
    }
    #pragma unroll
    for (int nt = 0; nt < 8; ++nt) {
      acc[nt] = __builtin_amdgcn_wmma_f32_16x16x32_bf16(
          /*neg_a=*/false, af, /*neg_b=*/false, bfrag[nt],
          /*c_mod=*/(short)0, acc[nt], /*reuse_a=*/false, /*reuse_b=*/false);
    }
    __syncthreads();
  }

  // ---- epilogue (C/D layout: VGPR i -> row i + kh*8, col = lane&15) ----
  float sg = 0.0f;
  if (epi == EPI_SKIP) sg = 1.0f / (1.0f + __expf(-skipgate[0]));
  #pragma unroll
  for (int nt = 0; nt < 8; ++nt) {
    const int col = n0 + nt * 16 + lh;
    #pragma unroll
    for (int i = 0; i < 8; ++i) {
      const int row = m0 + wave * 16 + kh * 8 + i;
      if (row < M && col < Nw) {
        float v = acc[nt][i] + bias[col];
        if (epi == EPI_RELU)      v = fmaxf(v, 0.0f);
        else if (epi == EPI_SKIP) v = sg * v + (1.0f - sg) * Xprev[(size_t)row * Nw + col];
        Cout[(size_t)row * Nw + col] = v;
      }
    }
  }
}

// ---------------- helpers ----------------
__global__ __launch_bounds__(256) void f32_to_bf16_k(
    const float* __restrict__ in, __bf16* __restrict__ out, long n)
{
  long i = (long)blockIdx.x * 256 + threadIdx.x;
  if (i < n) out[i] = (__bf16)in[i];
}

__global__ __launch_bounds__(256) void fill_f32_k(float* __restrict__ p, float v, long n)
{
  long i = (long)blockIdx.x * 256 + threadIdx.x;
  if (i < n) p[i] = v;
}

__device__ __forceinline__ void atomicMaxFloat(float* addr, float val)
{
  // sign-split trick; correct given init to -inf
  if (val >= 0.0f) atomicMax((int*)addr, __float_as_int(val));
  else             atomicMin((unsigned int*)addr, __float_as_uint(val));
}

// -------------------------------------------------------------------
// Edge pass 1: alpha[e,h] = (q[dst,h,:] . (k[src,h,:] @ Wk[h])) * p_rel[h]/sqrt(D)
// plus atomic segment-max into amax[dst,h].
// -------------------------------------------------------------------
__global__ __launch_bounds__(256) void edge_alpha_k(
    const float* __restrict__ kqv, const int* __restrict__ src, const int* __restrict__ dst,
    int E, int soff, int doff, const float* __restrict__ Wk, const float* __restrict__ prel,
    float* __restrict__ alpha, float* __restrict__ amax)
{
  __shared__ float sW[H_DIM * D_DIM * D_DIM];  // 16 KB
  for (int i = threadIdx.x; i < H_DIM * D_DIM * D_DIM; i += 256) sW[i] = Wk[i];
  __syncthreads();

  long idx = (long)blockIdx.x * 256 + threadIdx.x;
  if (idx >= (long)E * H_DIM) return;
  int e = (int)(idx >> 2);
  int h = (int)(idx & 3);
  int s = src[e] + soff;
  int d = dst[e] + doff;

  const float* kk = kqv + (size_t)s * (3 * C_DIM) + h * D_DIM;            // k block
  const float* qq = kqv + (size_t)d * (3 * C_DIM) + C_DIM + h * D_DIM;    // q block
  float q[D_DIM];
  #pragma unroll
  for (int f = 0; f < D_DIM; ++f) q[f] = qq[f];

  float a = 0.0f;
  for (int dd = 0; dd < D_DIM; ++dd) {
    const float* wrow = sW + (h * D_DIM + dd) * D_DIM;
    float t = 0.0f;
    #pragma unroll
    for (int f = 0; f < D_DIM; ++f) t += wrow[f] * q[f];
    a += kk[dd] * t;
  }
  a *= prel[h] * INV_SQRT_D;
  alpha[idx] = a;
  atomicMaxFloat(&amax[(size_t)d * H_DIM + h], a);
}

// Edge pass 2: ea = exp(alpha - amax[dst]); denom[dst] += ea (in place on alpha buf)
__global__ __launch_bounds__(256) void edge_exp_k(
    const int* __restrict__ dst, int E, int doff,
    float* __restrict__ ea, const float* __restrict__ amax, float* __restrict__ denom)
{
  long idx = (long)blockIdx.x * 256 + threadIdx.x;
  if (idx >= (long)E * H_DIM) return;
  int e = (int)(idx >> 2);
  int h = (int)(idx & 3);
  int d = dst[e] + doff;
  float v = __expf(ea[idx] - amax[(size_t)d * H_DIM + h]);
  ea[idx] = v;
  atomicAdd(&denom[(size_t)d * H_DIM + h], v);
}

// Edge pass 3: agg[dst, h*D+f] += (ea/denom) * (v[src,h,:] @ Wv[h])[f]
__global__ __launch_bounds__(256) void edge_agg_k(
    const float* __restrict__ kqv, const int* __restrict__ src, const int* __restrict__ dst,
    int E, int soff, int doff, const float* __restrict__ Wv,
    const float* __restrict__ ea, const float* __restrict__ denom, float* __restrict__ agg)
{
  __shared__ float sW[H_DIM * D_DIM * D_DIM];
  for (int i = threadIdx.x; i < H_DIM * D_DIM * D_DIM; i += 256) sW[i] = Wv[i];
  __syncthreads();

  long idx = (long)blockIdx.x * 256 + threadIdx.x;
  if (idx >= (long)E * H_DIM) return;
  int e = (int)(idx >> 2);
  int h = (int)(idx & 3);
  int s = src[e] + soff;
  int d = dst[e] + doff;

  float w = ea[idx] / fmaxf(denom[(size_t)d * H_DIM + h], 1e-16f);
  const float* vv = kqv + (size_t)s * (3 * C_DIM) + 2 * C_DIM + h * D_DIM;
  float vr[D_DIM];
  #pragma unroll
  for (int dd = 0; dd < D_DIM; ++dd) vr[dd] = vv[dd];

  for (int f = 0; f < D_DIM; ++f) {
    float t = 0.0f;
    #pragma unroll
    for (int dd = 0; dd < D_DIM; ++dd) t += vr[dd] * sW[(h * D_DIM + dd) * D_DIM + f];
    atomicAdd(&agg[(size_t)d * C_DIM + h * D_DIM + f], w * t);
  }
}

// -------------------------------------------------------------------
// Orchestration
// -------------------------------------------------------------------
extern "C" void kernel_launch(void* const* d_in, const int* in_sizes, int n_in,
                              void* d_out, int out_size, void* d_ws, size_t ws_size,
                              hipStream_t stream)
{
  (void)in_sizes; (void)n_in; (void)out_size; (void)ws_size;

  const float* xs[3]   = {(const float*)d_in[0], (const float*)d_in[1], (const float*)d_in[2]};
  const int*   esrc[4] = {(const int*)d_in[3], (const int*)d_in[5], (const int*)d_in[7], (const int*)d_in[9]};
  const int*   edst[4] = {(const int*)d_in[4], (const int*)d_in[6], (const int*)d_in[8], (const int*)d_in[10]};
  const float* W_in  = (const float*)d_in[11];
  const float* b_in  = (const float*)d_in[12];
  const float* Wkqv  = (const float*)d_in[13];
  const float* bkqv  = (const float*)d_in[14];
  const float* Wkrel = (const float*)d_in[15];
  const float* Wvrel = (const float*)d_in[16];
  const float* prel  = (const float*)d_in[17];
  const float* Wout  = (const float*)d_in[18];
  const float* bout  = (const float*)d_in[19];
  const float* skip  = (const float*)d_in[20];

  const int  Esz[4]  = {150000, 150000, 200000, 100000};
  const int  soff[4] = {NP, 0, 0, 0};           // src type offsets (writes: author)
  const int  doff[4] = {0, NP, 0, NP + NA};     // dst type offsets
  const long eoff[4] = {0, 150000, 300000, 500000};
  const int  Mt[3]   = {NP, NA, NKW};
  const long toff[3] = {0, NP, NP + NA};

  // ---- workspace carve ----
  float* Xa    = (float*)d_ws;
  float* Xb    = Xa  + (size_t)NTOT * C_DIM;
  float* KQV   = Xb  + (size_t)NTOT * C_DIM;
  float* AGG   = KQV + (size_t)NTOT * 3 * C_DIM;
  float* ALPHA = AGG + (size_t)NTOT * C_DIM;
  float* AMAX  = ALPHA + (size_t)ETOT * H_DIM;
  float* DEN   = AMAX + (size_t)NTOT * H_DIM;
  __bf16* bWin  = (__bf16*)(DEN + (size_t)NTOT * H_DIM);
  __bf16* bWkqv = bWin  + (size_t)3 * C_DIM * C_DIM;
  __bf16* bWout = bWkqv + (size_t)L_DIM * 3 * C_DIM * 3 * C_DIM;

  // ---- convert weights fp32 -> bf16 (every call; no cached state) ----
  {
    long n1 = (long)3 * C_DIM * C_DIM;
    long n2 = (long)L_DIM * 3 * C_DIM * 3 * C_DIM;
    long n3 = (long)L_DIM * 3 * C_DIM * C_DIM;
    f32_to_bf16_k<<<(unsigned)((n1 + 255) / 256), 256, 0, stream>>>(W_in,  bWin,  n1);
    f32_to_bf16_k<<<(unsigned)((n2 + 255) / 256), 256, 0, stream>>>(Wkqv,  bWkqv, n2);
    f32_to_bf16_k<<<(unsigned)((n3 + 255) / 256), 256, 0, stream>>>(Wout,  bWout, n3);
  }

  auto launch_gemm = [&](const float* A, const __bf16* W, const float* bias, float* Co,
                         const float* Xp, const float* sgp, int M, int Nw, int Kd,
                         int actA, int epi) {
    dim3 g((unsigned)((M + BM - 1) / BM), (unsigned)((Nw + BN - 1) / BN));
    gemm_bf16_wmma<<<g, 256, 0, stream>>>(A, W, bias, Co, Xp, sgp, M, Nw, Kd, actA, epi);
  };

  // ---- input per-type linear + ReLU -> Xa ----
  for (int t = 0; t < 3; ++t)
    launch_gemm(xs[t], bWin + (size_t)t * C_DIM * C_DIM, b_in + (size_t)t * C_DIM,
                Xa + toff[t] * C_DIM, nullptr, nullptr, Mt[t], C_DIM, C_DIM, 0, EPI_RELU);

  const float* Xcur = Xa;
  for (int l = 0; l < L_DIM; ++l) {
    float* Xnext = (l == L_DIM - 1) ? (float*)d_out : Xb;

    // KQV projection per type
    for (int t = 0; t < 3; ++t)
      launch_gemm(Xcur + toff[t] * C_DIM,
                  bWkqv + (size_t)(l * 3 + t) * C_DIM * 3 * C_DIM,
                  bkqv + (size_t)(l * 3 + t) * 3 * C_DIM,
                  KQV + toff[t] * 3 * C_DIM,
                  nullptr, nullptr, Mt[t], 3 * C_DIM, C_DIM, 0, EPI_BIAS);

    // init segment buffers
    {
      long nh = (long)NTOT * H_DIM, nc = (long)NTOT * C_DIM;
      fill_f32_k<<<(unsigned)((nh + 255) / 256), 256, 0, stream>>>(AMAX, -INFINITY, nh);
      fill_f32_k<<<(unsigned)((nh + 255) / 256), 256, 0, stream>>>(DEN, 0.0f, nh);
      fill_f32_k<<<(unsigned)((nc + 255) / 256), 256, 0, stream>>>(AGG, 0.0f, nc);
    }

    // edge pass 1: alpha + segment max
    for (int r = 0; r < R_DIM; ++r) {
      long nt = (long)Esz[r] * H_DIM;
      edge_alpha_k<<<(unsigned)((nt + 255) / 256), 256, 0, stream>>>(
          KQV, esrc[r], edst[r], Esz[r], soff[r], doff[r],
          Wkrel + (size_t)(l * R_DIM + r) * H_DIM * D_DIM * D_DIM,
          prel + (size_t)(l * R_DIM + r) * H_DIM,
          ALPHA + eoff[r] * H_DIM, AMAX);
    }
    // edge pass 2: exp + denom
    for (int r = 0; r < R_DIM; ++r) {
      long nt = (long)Esz[r] * H_DIM;
      edge_exp_k<<<(unsigned)((nt + 255) / 256), 256, 0, stream>>>(
          edst[r], Esz[r], doff[r], ALPHA + eoff[r] * H_DIM, AMAX, DEN);
    }
    // edge pass 3: weighted v_rel scatter
    for (int r = 0; r < R_DIM; ++r) {
      long nt = (long)Esz[r] * H_DIM;
      edge_agg_k<<<(unsigned)((nt + 255) / 256), 256, 0, stream>>>(
          KQV, esrc[r], edst[r], Esz[r], soff[r], doff[r],
          Wvrel + (size_t)(l * R_DIM + r) * H_DIM * D_DIM * D_DIM,
          ALPHA + eoff[r] * H_DIM, DEN, AGG);
    }

    // output projection per type: gelu(agg) @ Wout + bout, sigmoid-skip blend
    for (int t = 0; t < 3; ++t)
      launch_gemm(AGG + toff[t] * C_DIM,
                  bWout + (size_t)(l * 3 + t) * C_DIM * C_DIM,
                  bout + (size_t)(l * 3 + t) * C_DIM,
                  Xnext + toff[t] * C_DIM,
                  Xcur + toff[t] * C_DIM, skip + (size_t)(l * 3 + t),
                  Mt[t], C_DIM, C_DIM, 1, EPI_SKIP);

    Xcur = Xnext;
  }
}
